// LIFNetwork_8701603742198
// MI455X (gfx1250) — matrix operand order repaired and checked by reading, not compiled, not run on hardware
//
#include <hip/hip_runtime.h>

// CDNA5 / gfx1250 WMMA types
typedef __attribute__((ext_vector_type(16))) _Float16 v16h;
typedef __attribute__((ext_vector_type(8)))  float    v8f;
typedef __attribute__((ext_vector_type(4)))  unsigned v4u;

static __device__ inline unsigned pack_h2(float a, float b) {
    union { _Float16 h[2]; unsigned u; } x;
    x.h[0] = (_Float16)a;
    x.h[1] = (_Float16)b;
    return x.u;
}

union Frag { unsigned u[8]; v4u q[2]; v16h h; };

// One workgroup = 256 threads = 8 wave32s.
// Workgroup owns batch b = blockIdx.y and 128 neuron columns [n0, n0+128).
// Wave w owns 16 columns; v/cur state lives in registers of lanes 0..15.
__global__ void __launch_bounds__(256, 1)
lif_wmma_kernel(const float* __restrict__ initial_v,
                const float* __restrict__ initial_I,
                const float* __restrict__ inputs,       // (B,T,NI) 0/1 spikes
                const float* __restrict__ ffw,          // (N,NI)
                const float* __restrict__ E_v_th,
                const float* __restrict__ I_v_th,
                const int*   __restrict__ neuron_types, // (N,)
                float*       __restrict__ out,          // v (B,N) then cur (B,N)
                int B, int T, int N, int NI)
{
    constexpr float ALPHA = 0.9048374180359595f;  // exp(-0.1), applied to current
    constexpr float BETA  = 0.8187307530779818f;  // exp(-0.2), applied to voltage

    const int tid  = threadIdx.x;
    const int lane = tid & 31;    // wave32
    const int wave = tid >> 5;
    const int g    = lane >> 4;   // lane half-group (ISA fragment layouts)
    const int ln   = lane & 15;
    const int b    = blockIdx.y;
    const int n0   = blockIdx.x * 128;
    const int col  = n0 + wave * 16 + ln;

    const int KC = NI >> 5;       // K chunks of 32 per WMMA

    extern __shared__ char smem[];
    // pB: B fragments pre-swizzled: [kc][g][vgpr j][col 0..127] dwords (2 f16 each)
    unsigned* pB = (unsigned*)smem;                         // NI*64 dwords
    // pA: A fragments pre-swizzled: [kc][g][m 0..15][12] dwords (8 used + 4 pad)
    unsigned* pA = pB + (size_t)NI * 64;                    // NI*12 dwords
    float*    sD = (float*)(pA + (size_t)NI * 12);          // [8 waves][16][16] f32

    // ---- one-time: load + f32->f16 convert + swizzle Wff slice into LDS ----
    for (int c = 0; c < 128; ++c) {
        const float* wrow = ffw + (size_t)(n0 + c) * NI;
        for (int k = tid * 4; k < NI; k += 256 * 4) {       // coalesced along K
            float w0 = wrow[k], w1 = wrow[k + 1], w2 = wrow[k + 2], w3 = wrow[k + 3];
            int kc = k >> 5, gg = (k >> 4) & 1, j = (k >> 1) & 7;
            pB[(((kc * 2 + gg) * 8 + j) << 7) + c] = pack_h2(w0, w1);
            int k2 = k + 2;
            kc = k2 >> 5; gg = (k2 >> 4) & 1; j = (k2 >> 1) & 7;
            pB[(((kc * 2 + gg) * 8 + j) << 7) + c] = pack_h2(w2, w3);
        }
    }

    // ---- persistent per-neuron state (lanes 0..15 of each wave are live) ----
    float v   = initial_v[(size_t)b * N + col];
    float cur = initial_I[(size_t)b * N + col];
    const float vth = (neuron_types[col] == 1) ? E_v_th[0] : I_v_th[0];

    __syncthreads();

    float* sd = sD + wave * 256;
    const int pairs = NI >> 1;
    const int cidx  = wave * 16 + ln;

    for (int t0 = 0; t0 < T; t0 += 16) {
        // ---- stage 16 timesteps of spikes, f32->f16, pre-swizzled to A-fragment order ----
        for (int m = 0; m < 16; ++m) {
            const int t = t0 + m;
            for (int i2 = tid; i2 < pairs; i2 += 256) {     // coalesced global reads
                float x0 = 0.0f, x1 = 0.0f;
                if (t < T) {
                    const float* ip = inputs + ((size_t)b * T + t) * NI + i2 * 2;
                    x0 = ip[0]; x1 = ip[1];
                }
                const int kc = i2 >> 4;
                const int r  = i2 & 15;
                const int gg = (r >> 3) & 1;
                const int j  = r & 7;
                pA[((kc * 2 + gg) * 16 + m) * 12 + j] = pack_h2(x0, x1);
            }
        }
        __syncthreads();

        // prefetch next chunk's spikes into cache while we compute (global_prefetch_b8)
        if (t0 + 16 < T) {
            const char* nx = (const char*)(inputs + ((size_t)b * T + (t0 + 16)) * NI)
                             + (size_t)tid * 256;
            __builtin_prefetch(nx, 0, 1);
        }

        // ---- ff tile: D(16t x 16n) = A(16t x NI) * B(NI x 16n), f32 acc ----
        // 2-deep software pipeline: independent fragment regs per K-step so the
        // DS loads for step k+1 overlap the WMMA for step k (partial dscnt waits).
        v8f acc = {};
        int kc = 0;
        for (; kc + 2 <= KC; kc += 2) {
            Frag a0, b0, a1, b1;
            {
                const v4u* ap = (const v4u*)(pA + ((kc * 2 + g) * 16 + ln) * 12);
                a0.q[0] = ap[0]; a0.q[1] = ap[1];           // 2x ds_load_b128
                const unsigned* bp = pB + (((kc * 2 + g) * 8) << 7) + cidx;
#pragma unroll
                for (int j = 0; j < 8; ++j) b0.u[j] = bp[j << 7];
            }
            {
                const v4u* ap = (const v4u*)(pA + (((kc + 1) * 2 + g) * 16 + ln) * 12);
                a1.q[0] = ap[0]; a1.q[1] = ap[1];
                const unsigned* bp = pB + ((((kc + 1) * 2 + g) * 8) << 7) + cidx;
#pragma unroll
                for (int j = 0; j < 8; ++j) b1.u[j] = bp[j << 7];
            }
            acc = __builtin_amdgcn_wmma_f32_16x16x32_f16(
                false, a0.h, false, b0.h, (short)0, acc, false, false);
            acc = __builtin_amdgcn_wmma_f32_16x16x32_f16(
                false, a1.h, false, b1.h, (short)0, acc, false, false);
        }
        for (; kc < KC; ++kc) {                             // tail (KC odd)
            Frag a0, b0;
            const v4u* ap = (const v4u*)(pA + ((kc * 2 + g) * 16 + ln) * 12);
            a0.q[0] = ap[0]; a0.q[1] = ap[1];
            const unsigned* bp = pB + (((kc * 2 + g) * 8) << 7) + cidx;
#pragma unroll
            for (int j = 0; j < 8; ++j) b0.u[j] = bp[j << 7];
            acc = __builtin_amdgcn_wmma_f32_16x16x32_f16(
                false, a0.h, false, b0.h, (short)0, acc, false, false);
        }

        // ---- D fragment -> LDS (rows = timesteps, cols = neurons) ----
#pragma unroll
        for (int r = 0; r < 8; ++r) sd[(r + g * 8) * 16 + ln] = acc[r];
        __syncthreads();

        // ---- sequential LIF scan over the 16 staged timesteps ----
        if (lane < 16) {
#pragma unroll
            for (int m = 0; m < 16; ++m) {
                if (t0 + m < T) {
                    float ff = sd[m * 16 + ln];
                    cur = ALPHA * cur + ff;        // recurrent term is exactly 0 (s==0)
                    v   = BETA * v + cur;          // - s == 0
                    v   = (v >= vth) ? 0.0f : v;   // spike reset
                }
            }
        }
        // next iteration's __syncthreads (after staging) orders sD/pA reuse
    }

    if (lane < 16) {
        const size_t o = (size_t)b * N + col;
        out[o] = v;                                // final v
        out[(size_t)B * N + o] = cur;              // final cur
    }
}

extern "C" void kernel_launch(void* const* d_in, const int* in_sizes, int n_in,
                              void* d_out, int out_size, void* d_ws, size_t ws_size,
                              hipStream_t stream) {
    (void)n_in; (void)out_size; (void)d_ws; (void)ws_size;

    // Shapes from in_sizes (deterministic, no device reads):
    // [0]=B*N initial_v, [2]=B*T*NI inputs, [3]=N*N recurrent, [4]=N*NI ff weights
    int N = 1;
    while ((long long)N * N < (long long)in_sizes[3]) ++N;
    const int B  = in_sizes[0] / N;
    const int NI = in_sizes[4] / N;
    const int T  = in_sizes[2] / (B * NI);

    const size_t smem = ((size_t)NI * 64     // pB dwords
                         + (size_t)NI * 12)  // pA dwords (pre-swizzled, padded)
                            * 4
                        + (size_t)8 * 256 * 4;  // sD
    hipFuncSetAttribute(reinterpret_cast<const void*>(lif_wmma_kernel),
                        hipFuncAttributeMaxDynamicSharedMemorySize, (int)smem);

    dim3 grid(N / 128, B);
    lif_wmma_kernel<<<grid, 256, smem, stream>>>(
        (const float*)d_in[0],   // initial_v
        (const float*)d_in[1],   // initial_I
        (const float*)d_in[2],   // inputs
        (const float*)d_in[4],   // feedforward_weights (recurrent is dead code)
        (const float*)d_in[7],   // E_v_th
        (const float*)d_in[8],   // I_v_th
        (const int*)d_in[9],     // neuron_types
        (float*)d_out, B, T, N, NI);
}
